// MultiscaleEdgeLayer_59064390255196
// MI455X (gfx1250) — matrix operand order repaired and checked by reading, not compiled, not run on hardware
//
#include <hip/hip_runtime.h>

typedef _Float16 v16h __attribute__((ext_vector_type(16)));
typedef float    v8f  __attribute__((ext_vector_type(8)));
typedef float    v4f  __attribute__((ext_vector_type(4)));

__device__ __forceinline__ v8f wmma16(v16h a, v16h b, v8f c) {
  return __builtin_amdgcn_wmma_f32_16x16x32_f16(false, a, false, b, (short)0, c,
                                                false, false);
}

// ---- weight repack: row-major f32 [K,N] -> chunked f16 B-fragments ----
// chunk (kc,nc) = one 32x16 B tile; within chunk, lane l (0..31), slot s (0..15):
//   k = kc*32 + (l>=16 ? 8 : 0) + (s & 7) + (s >= 8 ? 16 : 0) ;  n = nc*16 + (l & 15)
// so a wave's B fragment for (kc,nc) is 32 contiguous bytes per lane.
__global__ __launch_bounds__(256) void pack_w(const float* __restrict__ W,
                                              _Float16* __restrict__ out,
                                              int K, int N) {
  int tid = blockIdx.x * 256 + threadIdx.x;
  if (tid >= K * N) return;
  int nCN   = N >> 4;
  int chunk = tid >> 9;
  int within = tid & 511;
  int lane = within >> 4;
  int s    = within & 15;
  int kc = chunk / nCN;
  int nc = chunk - kc * nCN;
  int k = kc * 32 + ((lane >> 4) << 3) + (s & 7) + ((s >> 3) << 4);
  int n = nc * 16 + (lane & 15);
  out[tid] = (_Float16)W[k * N + n];
}

__global__ __launch_bounds__(256) void zero_f4(float* __restrict__ p, long long n4) {
  long long i = (long long)blockIdx.x * 256 + threadIdx.x;
  if (i < n4) ((v4f*)p)[i] = (v4f){0.f, 0.f, 0.f, 0.f};
}

// ---- edge kernel: gather -> 288->256->256->128 MLP (WMMA f16) -> atomic scatter
__global__ __launch_bounds__(128)
void edge_kernel(const float* __restrict__ x, const long long* __restrict__ ei,
                 const float* __restrict__ ea,
                 const _Float16* __restrict__ W1p, const float* __restrict__ b1,
                 const _Float16* __restrict__ W2p, const float* __restrict__ b2,
                 const _Float16* __restrict__ W3p, const float* __restrict__ b3,
                 float* __restrict__ aggr, long long E) {
  __shared__ _Float16 sh[4][16 * 256];     // 32 KB: one 16x256 f16 tile per wave
  const int wave = threadIdx.x >> 5;
  const int lane = threadIdx.x & 31;
  const int hh   = lane >> 4;              // half-wave select
  const int nIdx = lane & 15;              // A-row / B,C,D-column for this lane
  _Float16* h = sh[wave];

  const long long e0 = ((long long)blockIdx.x * 4 + wave) * 16;
  long long eA = e0 + nIdx; if (eA >= E) eA = E - 1;
  const long long rowI = ei[eA];
  const long long colI = ei[E + eA];
  const float* srcR = x + rowI * 128;
  const float* srcC = x + colI * 128;
  const float* srcE = ea + eA * 32;

  // Build A fragments for K=288 (x[row] | x[col] | edge_attr), f32 -> f16
  v16h A[9];
#pragma unroll
  for (int kc = 0; kc < 9; ++kc) {
    const float* base = (kc < 4) ? (srcR + kc * 32)
                      : (kc < 8) ? (srcC + (kc - 4) * 32)
                                 : srcE;
    const float* p0 = base + hh * 8;
#pragma unroll
    for (int j = 0; j < 8; ++j) {
      A[kc][j]     = (_Float16)p0[j];       // K offsets  0..7 (+8 for hh=1)
      A[kc][8 + j] = (_Float16)p0[16 + j];  // K offsets 16..23 (+8 for hh=1)
    }
  }

  // layer1: 288 -> 256, relu, store f16 tile to LDS (row-major [16][256])
  for (int nt = 0; nt < 16; ++nt) {
    float bv = b1[nt * 16 + nIdx];
    v8f acc;
#pragma unroll
    for (int r = 0; r < 8; ++r) acc[r] = bv;
#pragma unroll
    for (int kc = 0; kc < 9; ++kc) {
      v16h B = *(const v16h*)(W1p + ((kc * 16 + nt) * 512 + lane * 16));
      acc = wmma16(A[kc], B, acc);
    }
#pragma unroll
    for (int r = 0; r < 8; ++r) {
      float v = acc[r] > 0.f ? acc[r] : 0.f;
      h[(hh * 8 + r) * 256 + nt * 16 + nIdx] = (_Float16)v;
    }
  }

  // A2 fragments from LDS (same-wave LDS ops are in-order; no barrier needed)
  v16h A2[8];
#pragma unroll
  for (int kc = 0; kc < 8; ++kc) {
    const _Float16* p = h + nIdx * 256 + kc * 32 + hh * 8;
#pragma unroll
    for (int j = 0; j < 8; ++j) { A2[kc][j] = p[j]; A2[kc][8 + j] = p[16 + j]; }
  }

  // layer2: 256 -> 256, relu, back to LDS
  for (int nt = 0; nt < 16; ++nt) {
    float bv = b2[nt * 16 + nIdx];
    v8f acc;
#pragma unroll
    for (int r = 0; r < 8; ++r) acc[r] = bv;
#pragma unroll
    for (int kc = 0; kc < 8; ++kc) {
      v16h B = *(const v16h*)(W2p + ((kc * 16 + nt) * 512 + lane * 16));
      acc = wmma16(A2[kc], B, acc);
    }
#pragma unroll
    for (int r = 0; r < 8; ++r) {
      float v = acc[r] > 0.f ? acc[r] : 0.f;
      h[(hh * 8 + r) * 256 + nt * 16 + nIdx] = (_Float16)v;
    }
  }

  v16h A3[8];
#pragma unroll
  for (int kc = 0; kc < 8; ++kc) {
    const _Float16* p = h + nIdx * 256 + kc * 32 + hh * 8;
#pragma unroll
    for (int j = 0; j < 8; ++j) { A3[kc][j] = p[j]; A3[kc][8 + j] = p[16 + j]; }
  }

  // scatter destinations for the rows this lane holds in D tiles
  long long dst[8];
  bool live[8];
#pragma unroll
  for (int r = 0; r < 8; ++r) {
    long long er = e0 + hh * 8 + r;
    live[r] = er < E;
    dst[r] = ei[E + (live[r] ? er : 0)] * 128;
  }

  // layer3: 256 -> 128, scatter-add to aggr
  for (int nt = 0; nt < 8; ++nt) {
    float bv = b3[nt * 16 + nIdx];
    v8f acc;
#pragma unroll
    for (int r = 0; r < 8; ++r) acc[r] = bv;
#pragma unroll
    for (int kc = 0; kc < 8; ++kc) {
      v16h B = *(const v16h*)(W3p + ((kc * 8 + nt) * 512 + lane * 16));
      acc = wmma16(A3[kc], B, acc);
    }
#pragma unroll
    for (int r = 0; r < 8; ++r) {
      if (live[r])
        __hip_atomic_fetch_add(&aggr[dst[r] + nt * 16 + nIdx], acc[r],
                               __ATOMIC_RELAXED, __HIP_MEMORY_SCOPE_AGENT);
    }
  }
}

// ---- node kernel: concat(x, aggr) -> 256->256->128 MLP + residual -> y (d_out)
__global__ __launch_bounds__(128)
void node_kernel(const float* __restrict__ x, const float* __restrict__ aggr,
                 const _Float16* __restrict__ Wn1p, const float* __restrict__ bn1,
                 const _Float16* __restrict__ Wn2p, const float* __restrict__ bn2,
                 float* __restrict__ y, int N) {
  __shared__ _Float16 sh[4][16 * 256];
  const int wave = threadIdx.x >> 5;
  const int lane = threadIdx.x & 31;
  const int hh   = lane >> 4;
  const int nIdx = lane & 15;
  _Float16* h = sh[wave];

  const int n0 = (blockIdx.x * 4 + wave) * 16;
  int na = n0 + nIdx; if (na >= N) na = N - 1;
  const float* srcX = x + (long long)na * 128;
  const float* srcA = aggr + (long long)na * 128;

  v16h A[8];
#pragma unroll
  for (int kc = 0; kc < 8; ++kc) {
    const float* base = (kc < 4) ? (srcX + kc * 32) : (srcA + (kc - 4) * 32);
    const float* p0 = base + hh * 8;
#pragma unroll
    for (int j = 0; j < 8; ++j) {
      A[kc][j]     = (_Float16)p0[j];
      A[kc][8 + j] = (_Float16)p0[16 + j];
    }
  }

  for (int nt = 0; nt < 16; ++nt) {
    float bv = bn1[nt * 16 + nIdx];
    v8f acc;
#pragma unroll
    for (int r = 0; r < 8; ++r) acc[r] = bv;
#pragma unroll
    for (int kc = 0; kc < 8; ++kc) {
      v16h B = *(const v16h*)(Wn1p + ((kc * 16 + nt) * 512 + lane * 16));
      acc = wmma16(A[kc], B, acc);
    }
#pragma unroll
    for (int r = 0; r < 8; ++r) {
      float v = acc[r] > 0.f ? acc[r] : 0.f;
      h[(hh * 8 + r) * 256 + nt * 16 + nIdx] = (_Float16)v;
    }
  }

  v16h A2[8];
#pragma unroll
  for (int kc = 0; kc < 8; ++kc) {
    const _Float16* p = h + nIdx * 256 + kc * 32 + hh * 8;
#pragma unroll
    for (int j = 0; j < 8; ++j) { A2[kc][j] = p[j]; A2[kc][8 + j] = p[16 + j]; }
  }

  for (int nt = 0; nt < 8; ++nt) {
    float bv = bn2[nt * 16 + nIdx];
    v8f acc;
#pragma unroll
    for (int r = 0; r < 8; ++r) acc[r] = bv;
#pragma unroll
    for (int kc = 0; kc < 8; ++kc) {
      v16h B = *(const v16h*)(Wn2p + ((kc * 8 + nt) * 512 + lane * 16));
      acc = wmma16(A2[kc], B, acc);
    }
#pragma unroll
    for (int r = 0; r < 8; ++r) {
      int nr = n0 + hh * 8 + r;
      if (nr < N) {
        long long off = (long long)nr * 128 + nt * 16 + nIdx;
        y[off] = x[off] + acc[r];   // residual
      }
    }
  }
}

// ---- LayerNorm over 128 features, one wave per node, in-place on d_out ----
__global__ __launch_bounds__(256)
void ln_kernel(float* __restrict__ y, const float* __restrict__ gamma,
               const float* __restrict__ beta, int N) {
  const int wave = threadIdx.x >> 5;
  const int lane = threadIdx.x & 31;
  long long node = (long long)blockIdx.x * 8 + wave;
  if (node >= N) return;
  float* yr = y + node * 128;
  v4f v = *(const v4f*)(yr + lane * 4);
  float s  = v[0] + v[1] + v[2] + v[3];
  float s2 = v[0] * v[0] + v[1] * v[1] + v[2] * v[2] + v[3] * v[3];
#pragma unroll
  for (int off = 16; off > 0; off >>= 1) {
    s  += __shfl_xor(s, off, 32);
    s2 += __shfl_xor(s2, off, 32);
  }
  float mean = s * (1.f / 128.f);
  float var  = s2 * (1.f / 128.f) - mean * mean;
  float inv  = rsqrtf(var + 1e-5f);
  v4f g = *(const v4f*)(gamma + lane * 4);
  v4f b = *(const v4f*)(beta + lane * 4);
  v4f o;
#pragma unroll
  for (int i = 0; i < 4; ++i) o[i] = (v[i] - mean) * inv * g[i] + b[i];
  *(v4f*)(yr + lane * 4) = o;
}

extern "C" void kernel_launch(void* const* d_in, const int* in_sizes, int n_in,
                              void* d_out, int out_size, void* d_ws, size_t ws_size,
                              hipStream_t stream) {
  (void)n_in; (void)out_size; (void)ws_size;
  const float*     x   = (const float*)d_in[0];
  const long long* ei  = (const long long*)d_in[1];
  const float*     ea  = (const float*)d_in[2];
  const float* W1  = (const float*)d_in[3];
  const float* b1  = (const float*)d_in[4];
  const float* W2  = (const float*)d_in[5];
  const float* b2  = (const float*)d_in[6];
  const float* W3  = (const float*)d_in[7];
  const float* b3  = (const float*)d_in[8];
  const float* Wn1 = (const float*)d_in[9];
  const float* bn1 = (const float*)d_in[10];
  const float* Wn2 = (const float*)d_in[11];
  const float* bn2 = (const float*)d_in[12];
  const float* gamma = (const float*)d_in[13];
  const float* beta  = (const float*)d_in[14];

  const int       N = in_sizes[0] / 128;
  const long long E = in_sizes[1] / 2;

  // workspace layout: aggr (N*128 f32) | packed f16 weights
  char* ws = (char*)d_ws;
  float* aggr = (float*)ws;
  size_t off = ((size_t)N * 128 * sizeof(float) + 255) & ~(size_t)255;
  _Float16* W1p  = (_Float16*)(ws + off); off += 288 * 256 * 2;
  _Float16* W2p  = (_Float16*)(ws + off); off += 256 * 256 * 2;
  _Float16* W3p  = (_Float16*)(ws + off); off += 256 * 128 * 2;
  _Float16* Wn1p = (_Float16*)(ws + off); off += 256 * 256 * 2;
  _Float16* Wn2p = (_Float16*)(ws + off); off += 256 * 128 * 2;

  pack_w<<<(288 * 256 + 255) / 256, 256, 0, stream>>>(W1, W1p, 288, 256);
  pack_w<<<(256 * 256 + 255) / 256, 256, 0, stream>>>(W2, W2p, 256, 256);
  pack_w<<<(256 * 128 + 255) / 256, 256, 0, stream>>>(W3, W3p, 256, 128);
  pack_w<<<(256 * 256 + 255) / 256, 256, 0, stream>>>(Wn1, Wn1p, 256, 256);
  pack_w<<<(256 * 128 + 255) / 256, 256, 0, stream>>>(Wn2, Wn2p, 256, 128);

  long long n4 = (long long)N * 128 / 4;
  zero_f4<<<(int)((n4 + 255) / 256), 256, 0, stream>>>(aggr, n4);

  int eBlocks = (int)((E + 63) / 64);   // 64 edges per 128-thread block
  edge_kernel<<<eBlocks, 128, 0, stream>>>(x, ei, ea, W1p, b1, W2p, b2, W3p, b3,
                                           aggr, E);

  int nBlocks = (N + 63) / 64;
  node_kernel<<<nBlocks, 128, 0, stream>>>(x, aggr, Wn1p, bn1, Wn2p, bn2,
                                           (float*)d_out, N);

  ln_kernel<<<(N + 7) / 8, 256, 0, stream>>>((float*)d_out, gamma, beta, N);
}
